// GATLayer_45629732553105
// MI455X (gfx1250) — matrix operand order repaired and checked by reading, not compiled, not run on hardware
//
#include <hip/hip_runtime.h>
#include <hip/hip_bf16.h>

// GAT layer for MI455X (gfx1250, wave32). bf16 WMMA (f32 accum) for all GEMMs;
// memory-bound pipeline otherwise. Weights pre-converted to bf16 once per launch.

typedef __attribute__((ext_vector_type(16))) __bf16 v16bf;
typedef __attribute__((ext_vector_type(8)))  float  v8f;

#define DN 128
#define DE 64
#define HF 64   // H*F
#define NH 4    // heads
#define FF 16   // feats per head

// ---- order-preserving float <-> uint for atomic segment-max ----
__device__ __forceinline__ unsigned f2ord(float f) {
  unsigned b = __float_as_uint(f);
  return (b & 0x80000000u) ? ~b : (b | 0x80000000u);
}
__device__ __forceinline__ float ord2f(unsigned k) {
  unsigned b = (k & 0x80000000u) ? (k ^ 0x80000000u) : ~k;
  return __uint_as_float(b);
}
__device__ __forceinline__ float leaky(float x) { return x > 0.f ? x : 0.01f * x; }

// A-matrix (16-bit, 16x32, contiguous k order, same as load_b16q): lane m = lane&15, half = lane>>4.
// elems 0..7  -> A[m, kbase + 16*half + j]
// elems 8..15 -> A[m, kbase + 16*half + 8 + j]
__device__ __forceinline__ v16bf load_a16(const float* rowp, int kbase, int half) {
  v16bf a;
  const float* p0 = rowp + kbase + 16 * half;
  const float* p1 = p0 + 8;
#pragma unroll
  for (int j = 0; j < 8; ++j) { a[j] = (__bf16)p0[j]; a[8 + j] = (__bf16)p1[j]; }
  return a;
}

// B-matrix (16-bit, 32x16): lane n = lane&15; halves split K 0-15 / 16-31.
// elem j -> B[kbase + 16*half + j][n]; B[k][n] = W[n*ldk + k], W pre-converted bf16.
// 16 contiguous bf16 = 32 bytes, 32B-aligned -> single vector load.
__device__ __forceinline__ v16bf load_b16q(const __bf16* __restrict__ W, int n, int ldk,
                                           int kbase, int half) {
  return *(const v16bf*)(W + (size_t)n * ldk + kbase + 16 * half);
}

__device__ __forceinline__ v8f wmma_bf16(v16bf a, v16bf b, v8f c) {
  return __builtin_amdgcn_wmma_f32_16x16x32_bf16(false, a, false, b, (short)0, c,
                                                 false, false);
}

// ---------------- K0: init accumulators ----------------
__global__ void k_init(float* __restrict__ nout, unsigned* __restrict__ mkey,
                       float* __restrict__ ssum, long nOut, long nSeg) {
  long i = (long)blockIdx.x * blockDim.x + threadIdx.x;
  if (i < nOut) nout[i] = 0.f;
  if (i < nSeg) { mkey[i] = 0u; ssum[i] = 0.f; }
}

// ---------------- K0b: weight f32 -> bf16 (launch-invariant, tiny) ----------------
__global__ void k_cvt_w(const float* __restrict__ Wn, const float* __restrict__ We,
                        const float* __restrict__ ew, __bf16* __restrict__ wnB,
                        __bf16* __restrict__ weB, __bf16* __restrict__ ewB) {
  int i = blockIdx.x * blockDim.x + threadIdx.x;
  if (i < DN * HF) wnB[i] = (__bf16)Wn[i];
  if (i < DE * HF) weB[i] = (__bf16)We[i];
  if (i < HF * HF) ewB[i] = (__bf16)ew[i];
}

// ---------------- K1: z = nf @ Wn^T  [N,128]x[128,64] ----------------
__global__ void k_node_proj(const float* __restrict__ nf, const __bf16* __restrict__ wnB,
                            float* __restrict__ z, int nTiles) {
  int lane = threadIdx.x & 31;
  int wave = threadIdx.x >> 5;
  int tile = blockIdx.x * (blockDim.x >> 5) + wave;
  if (tile >= nTiles) return;                 // wave-uniform: EXEC stays full
  int m15 = lane & 15, half = lane >> 4;
  const float* arow = nf + ((size_t)tile * 16 + m15) * DN;
  v8f acc[4] = {};
#pragma unroll
  for (int ks = 0; ks < DN / 32; ++ks) {
    v16bf a = load_a16(arow, ks * 32, half);
#pragma unroll
    for (int nt = 0; nt < 4; ++nt) {
      v16bf b = load_b16q(wnB, nt * 16 + m15, DN, ks * 32, half);
      acc[nt] = wmma_bf16(a, b, acc[nt]);
    }
  }
  // D layout: elem r -> row (r + 8*half), col m15
  size_t rbase = (size_t)tile * 16 + 8 * half;
#pragma unroll
  for (int nt = 0; nt < 4; ++nt)
#pragma unroll
    for (int r = 0; r < 8; ++r)
      z[(rbase + r) * HF + nt * 16 + m15] = acc[nt][r];
}

// ---------------- K1b: per-node attention dots ----------------
__global__ void k_node_attn(const float* __restrict__ z, const float* __restrict__ attn_l,
                            const float* __restrict__ attn_r, float* __restrict__ el,
                            float* __restrict__ er, int N) {
  int n = blockIdx.x * blockDim.x + threadIdx.x;
  if (n >= N) return;
  const float* zr = z + (size_t)n * HF;
#pragma unroll
  for (int h = 0; h < NH; ++h) {
    float sl = 0.f, sr = 0.f;
#pragma unroll
    for (int f = 0; f < FF; ++f) {
      float v = zr[h * FF + f];
      sl += v * attn_l[h * FF + f];
      sr += v * attn_r[h * FF + f];
    }
    el[(size_t)n * NH + h] = sl;
    er[(size_t)n * NH + h] = sr;
  }
}

// ---------------- K2: edge GEMMs + fused output + logits + seg-max ----------------
__global__ void k_edge_main(const float* __restrict__ ef, const int* __restrict__ src,
                            const int* __restrict__ dst, const __bf16* __restrict__ weB,
                            const __bf16* __restrict__ ewB, const float* __restrict__ attn_e,
                            const float* __restrict__ bias, const float* __restrict__ z,
                            const float* __restrict__ el, const float* __restrict__ er,
                            float* __restrict__ fusedOut, float* __restrict__ logits,
                            unsigned* __restrict__ mkey, int nTiles) {
  __shared__ float lds[8][16 * HF];          // 4 KB per wave, wave-private slice
  int lane = threadIdx.x & 31;
  int wave = threadIdx.x >> 5;
  int tile = blockIdx.x * (blockDim.x >> 5) + wave;
  if (tile >= nTiles) return;                // wave-uniform exit
  int m15 = lane & 15, half = lane >> 4;
  long e0 = (long)tile * 16;
  float* sl = lds[wave];

  // GEMM A: z_e = ef @ We^T  (16x64 tile of edges)
  const float* arow = ef + (e0 + m15) * DE;
  v16bf a0 = load_a16(arow, 0, half);
  v16bf a1 = load_a16(arow, 32, half);
#pragma unroll
  for (int nt = 0; nt < 4; ++nt) {
    v8f acc = {};
    acc = wmma_bf16(a0, load_b16q(weB, nt * 16 + m15, DE, 0, half), acc);
    acc = wmma_bf16(a1, load_b16q(weB, nt * 16 + m15, DE, 32, half), acc);
#pragma unroll
    for (int r = 0; r < 8; ++r)
      sl[(r + 8 * half) * HF + nt * 16 + m15] = acc[r];   // z_e -> LDS (f32)
  }

  // fused = z[src] + z_e + bias  (wave covers 16*64 elems; L2-resident z gather)
#pragma unroll 4
  for (int i = 0; i < 32; ++i) {
    int idx = i * 32 + lane;
    int le = idx >> 6, col = idx & 63;
    int sn = src[e0 + le];
    fusedOut[(e0 + le) * HF + col] = z[(size_t)sn * HF + col] + sl[idx] + bias[col];
  }

  // GEMM B: wze = z_e @ ew^T  (A re-read from LDS in A-layout, converted to bf16)
  v16bf c0 = load_a16(sl + m15 * HF, 0, half);
  v16bf c1 = load_a16(sl + m15 * HF, 32, half);
  v8f wacc[4];
#pragma unroll
  for (int nt = 0; nt < 4; ++nt) {
    v8f acc = {};
    acc = wmma_bf16(c0, load_b16q(ewB, nt * 16 + m15, HF, 0, half), acc);
    acc = wmma_bf16(c1, load_b16q(ewB, nt * 16 + m15, HF, 32, half), acc);
    wacc[nt] = acc;
  }
#pragma unroll
  for (int nt = 0; nt < 4; ++nt)
#pragma unroll
    for (int r = 0; r < 8; ++r)
      sl[(r + 8 * half) * HF + nt * 16 + m15] = wacc[nt][r];  // wze -> LDS

  // logits + segment max (divergent tail: all WMMAs already issued)
  if (lane < 16) {
    long e = e0 + lane;
    int sn = src[e], dn = dst[e];
    const float* wr = sl + lane * HF;
#pragma unroll
    for (int h = 0; h < NH; ++h) {
      float ee = 0.f;
#pragma unroll
      for (int f = 0; f < FF; ++f) ee += wr[h * FF + f] * attn_e[h * FF + f];
      float lg = leaky(el[(size_t)sn * NH + h] + er[(size_t)dn * NH + h] + ee);
      logits[e * NH + h] = lg;
      atomicMax(&mkey[(size_t)dn * NH + h], f2ord(lg));
    }
  }
}

// ---------------- K3: exp + segment sum ----------------
__global__ void k_softmax_num(const int* __restrict__ dst, const unsigned* __restrict__ mkey,
                              float* __restrict__ logits, float* __restrict__ ssum, long EH) {
  long i = (long)blockIdx.x * blockDim.x + threadIdx.x;
  if (i >= EH) return;
  long e = i >> 2;
  int h = (int)(i & 3);
  int d = dst[e];
  float m = ord2f(mkey[(size_t)d * NH + h]);
  float ex = expf(logits[i] - m);
  logits[i] = ex;                       // reuse buffer as softmax numerator
  atomicAdd(&ssum[(size_t)d * NH + h], ex);
}

// ---------------- K4: alpha-weighted aggregation into n_out ----------------
__global__ void k_aggregate(const int* __restrict__ src, const int* __restrict__ dst,
                            const float* __restrict__ exv, const float* __restrict__ ssum,
                            const float* __restrict__ z, float* __restrict__ nout, long total) {
  long i = (long)blockIdx.x * blockDim.x + threadIdx.x;
  if (i >= total) return;
  long e = i >> 6;
  int col = (int)(i & 63), h = col >> 4;
  int sn = src[e], dn = dst[e];
  float alpha = exv[e * NH + h] / ssum[(size_t)dn * NH + h];
  atomicAdd(&nout[(size_t)dn * HF + col], alpha * z[(size_t)sn * HF + col]);
}

extern "C" void kernel_launch(void* const* d_in, const int* in_sizes, int n_in,
                              void* d_out, int out_size, void* d_ws, size_t ws_size,
                              hipStream_t stream) {
  const float* nf     = (const float*)d_in[0];
  const float* ef     = (const float*)d_in[1];
  const int*   src    = (const int*)d_in[2];
  const int*   dst    = (const int*)d_in[3];
  const float* Wn     = (const float*)d_in[4];
  const float* We     = (const float*)d_in[5];
  const float* attn_l = (const float*)d_in[6];
  const float* attn_r = (const float*)d_in[7];
  const float* attn_e = (const float*)d_in[8];
  const float* ew     = (const float*)d_in[9];
  const float* bias   = (const float*)d_in[10];

  long N = in_sizes[0] / DN;     // 50000  (multiple of 16)
  long E = in_sizes[1] / DE;     // 800000 (multiple of 16)

  float* nout  = (float*)d_out;          // [N,64]
  float* fused = nout + (size_t)N * HF;  // [E,64]

  // workspace carve-up (~29 MB + 32 KB bf16 weights); every region offset is
  // a multiple of 32 bytes for these sizes (N*4 and E*4 floats are 32B-multiples)
  float* ws = (float*)d_ws;
  float*    z      = ws;              ws += (size_t)N * HF;
  float*    el     = ws;              ws += (size_t)N * NH;
  float*    er     = ws;              ws += (size_t)N * NH;
  float*    logits = ws;              ws += (size_t)E * NH;
  unsigned* mkey   = (unsigned*)ws;   ws += (size_t)N * NH;
  float*    ssum   = ws;              ws += (size_t)N * NH;
  __bf16*   wnB    = (__bf16*)ws;     // DN*HF = 8192 bf16
  __bf16*   weB    = wnB + DN * HF;   // DE*HF = 4096 bf16
  __bf16*   ewB    = weB + DE * HF;   // HF*HF = 4096 bf16

  long nOut = N * (long)HF;
  k_init<<<(int)((nOut + 255) / 256), 256, 0, stream>>>(nout, mkey, ssum, nOut, N * NH);

  k_cvt_w<<<(DN * HF + 255) / 256, 256, 0, stream>>>(Wn, We, ew, wnB, weB, ewB);

  int nTilesN = (int)(N / 16);
  k_node_proj<<<(nTilesN + 3) / 4, 128, 0, stream>>>(nf, wnB, z, nTilesN);

  k_node_attn<<<(int)((N + 255) / 256), 256, 0, stream>>>(z, attn_l, attn_r, el, er, (int)N);

  int nTilesE = (int)(E / 16);
  k_edge_main<<<(nTilesE + 7) / 8, 256, 0, stream>>>(ef, src, dst, weB, ewB, attn_e, bias,
                                                     z, el, er, fused, logits, mkey, nTilesE);

  long EH = E * (long)NH;
  k_softmax_num<<<(int)((EH + 255) / 256), 256, 0, stream>>>(dst, mkey, logits, ssum, EH);

  long tot = E * (long)HF;
  k_aggregate<<<(int)((tot + 255) / 256), 256, 0, stream>>>(src, dst, logits, ssum, z, nout, tot);
}
